// DeltaTSBlock_36910948942192
// MI455X (gfx1250) — compile-verified, matrix operand order checked by reading, not run on hardware
//
#include <hip/hip_runtime.h>

// ---------------------------------------------------------------------------
// Problem constants (B=4, T=4096, D=1024, H=16, d=64)
// ---------------------------------------------------------------------------
#define BB      4
#define T_LEN   4096
#define DMODEL  1024
#define HEADS   16
#define DHEAD   64
#define NROWS   (BB * T_LEN)        // 16384
#define CHUNK   16
#define NCHUNK  (T_LEN / CHUNK)     // 256
#define LSTRIDE 68                  // LDS row stride (floats): 16B-multiple rows
#define TSTRIDE 20                  // 16x16 LDS tiles row stride

typedef __attribute__((ext_vector_type(2)))  float     v2f;
typedef __attribute__((ext_vector_type(8)))  float     v8f;
typedef __attribute__((ext_vector_type(4)))  _Float16  v4h;
typedef __attribute__((ext_vector_type(8)))  _Float16  v8h;
typedef __attribute__((ext_vector_type(16))) _Float16  v16h;

// f32 WMMA 16x16x4: exact fp32 path (recurrence math)
__device__ __forceinline__ v8f wmma_f32(v2f a, v2f b, v8f c) {
  return __builtin_amdgcn_wmma_f32_16x16x4_f32(false, a, false, b,
                                               (short)0, c, false, false);
}
// f16 WMMA 16x16x32: fast path for the big GEMMs (f32 accumulate)
__device__ __forceinline__ v8f wmma_f16(v16h a, v16h b, v8f c) {
  return __builtin_amdgcn_wmma_f32_16x16x32_f16(false, a, false, b,
                                                (short)0, c, false, false);
}

// gfx1250 async global->LDS copies (ASYNCcnt) -------------------------------
__device__ __forceinline__ void async_copy_b128(unsigned lds_byte_off,
                                                const float* gaddr) {
  asm volatile("global_load_async_to_lds_b128 %0, %1, off"
               :: "v"(lds_byte_off), "v"(gaddr) : "memory");
}
__device__ __forceinline__ void async_copy_b32(unsigned lds_byte_off,
                                               const float* gaddr) {
  asm volatile("global_load_async_to_lds_b32 %0, %1, off"
               :: "v"(lds_byte_off), "v"(gaddr) : "memory");
}
__device__ __forceinline__ void wait_async() {
  asm volatile("s_wait_asynccnt 0" ::: "memory");
}

// ---------------------------------------------------------------------------
// Kernel 0: f32 -> f16 cast (vectorized x4)
// ---------------------------------------------------------------------------
__global__ __launch_bounds__(256) void cast_f16_kernel(
    const float* __restrict__ s, _Float16* __restrict__ d, long n4)
{
  const long i = (long)blockIdx.x * 256 + threadIdx.x;
  if (i < n4) {
    const float4 f = ((const float4*)s)[i];
    v4h o;
    o.x = (_Float16)f.x; o.y = (_Float16)f.y;
    o.z = (_Float16)f.z; o.w = (_Float16)f.w;
    ((v4h*)d)[i] = o;
  }
}

// ---------------------------------------------------------------------------
// Kernel 1: prep — rk = L2-normalize per-head x slices; gate/beta projections.
// ---------------------------------------------------------------------------
__global__ __launch_bounds__(512) void prep_kernel(
    const float* __restrict__ x,
    const float* __restrict__ Wg, const float* __restrict__ Wb,
    float* __restrict__ rk, float* __restrict__ gateO, float* __restrict__ betaO)
{
  const long n   = blockIdx.x;
  const int  h   = threadIdx.x >> 5;
  const int  ln  = threadIdx.x & 31;
  const float* xr = x + n * (long)DMODEL;

  float a0 = xr[h * DHEAD + ln * 2 + 0];
  float a1 = xr[h * DHEAD + ln * 2 + 1];
  float ss = a0 * a0 + a1 * a1;
  #pragma unroll
  for (int o = 16; o > 0; o >>= 1) ss += __shfl_xor(ss, o, 32);
  float inv = 1.0f / fmaxf(sqrtf(ss), 1e-12f);
  rk[n * (long)DMODEL + h * DHEAD + ln * 2 + 0] = a0 * inv;
  rk[n * (long)DMODEL + h * DHEAD + ln * 2 + 1] = a1 * inv;

  const float* wg = Wg + h * (long)DMODEL;
  const float* wb = Wb + h * (long)DMODEL;
  float dg = 0.f, db = 0.f;
  #pragma unroll 4
  for (int j = ln; j < DMODEL; j += 32) {
    float xv = xr[j];
    dg = fmaf(xv, wg[j], dg);
    db = fmaf(xv, wb[j], db);
  }
  #pragma unroll
  for (int o = 16; o > 0; o >>= 1) {
    dg += __shfl_xor(dg, o, 32);
    db += __shfl_xor(db, o, 32);
  }
  if (ln == 0) {
    gateO[n * HEADS + h] = 1.f / (1.f + expf(-dg));
    betaO[n * HEADS + h] = 1.f / (1.f + expf(-db));
  }
}

// ---------------------------------------------------------------------------
// Kernel 2: per-chunk precompute (input-only, fully parallel, all-WMMA):
//   Aw = W*W^T ;  Tm = (I + diag(beta)*strict_lower(Aw))^{-1} diag(beta)
//   Mm = lower_incl(R*W^T)
// One wave per (b,h,chunk) -> 16384 blocks.
// ---------------------------------------------------------------------------
__global__ __launch_bounds__(32) void chunk_precompute(
    const float* __restrict__ rk, const float* __restrict__ beta,
    float* __restrict__ TmG, float* __restrict__ MmG)
{
  __shared__ float Wc[CHUNK][LSTRIDE];
  __shared__ float Rc[CHUNK][LSTRIDE];
  __shared__ float Aw[16][TSTRIDE];
  __shared__ float betaS[16];

  const int idx = blockIdx.x;              // bh*NCHUNK + ci
  const int bh  = idx / NCHUNK, ci = idx % NCHUNK;
  const int b   = bh / HEADS,   h  = bh % HEADS;
  const int t0  = ci * CHUNK;
  const long rowBase = (long)b * T_LEN;
  const int ln = threadIdx.x;
  const int nH = ln & 15, hi = ln >> 4;

  for (int i = ln; i < CHUNK * DHEAD; i += 32) {
    const int r = i >> 6, c = i & 63;
    Rc[r][c] = rk[(rowBase + t0 + r) * (long)DMODEL + h * DHEAD + c];
    const int tw = t0 + r - 1;             // wk[t] = rk[t-1], wk[0] = 0
    Wc[r][c] = (tw >= 0) ? rk[(rowBase + tw) * (long)DMODEL + h * DHEAD + c]
                         : 0.f;
  }
  if (ln < 16) betaS[ln] = beta[(rowBase + t0 + ln) * HEADS + h];
  __syncthreads();

  {  // Aw = W * W^T
    v8f acc = {};
    #pragma unroll
    for (int kk = 0; kk < 16; ++kk) {
      const int kb = kk * 4;
      v2f a, bf;
      a.x  = Wc[nH][kb + hi * 2 + 0];
      a.y  = Wc[nH][kb + hi * 2 + 1];
      bf.x = Wc[nH][kb + hi * 2 + 0];
      bf.y = Wc[nH][kb + hi * 2 + 1];
      acc = wmma_f32(a, bf, acc);
    }
    const int rb = hi * 8;
    #pragma unroll
    for (int g = 0; g < 8; ++g) Aw[rb + g][nH] = acc[g];
  }
  {  // Mm = lower_incl(R * W^T), premasked, straight to global
    v8f acc = {};
    #pragma unroll
    for (int kk = 0; kk < 16; ++kk) {
      const int kb = kk * 4;
      v2f a, bf;
      a.x  = Rc[nH][kb + hi * 2 + 0];
      a.y  = Rc[nH][kb + hi * 2 + 1];
      bf.x = Wc[nH][kb + hi * 2 + 0];
      bf.y = Wc[nH][kb + hi * 2 + 1];
      acc = wmma_f32(a, bf, acc);
    }
    float* mm = MmG + (size_t)idx * 256;
    #pragma unroll
    for (int g = 0; g < 8; ++g) {
      const int r = hi * 8 + g;
      mm[r * 16 + nH] = (nH <= r) ? acc[g] : 0.f;
    }
  }
  __syncthreads();

  if (ln < 16) {                            // Tm forward substitution, lane=col
    float y[16];
    #pragma unroll
    for (int i = 0; i < 16; ++i) {
      float s = (i == ln) ? 1.0f : 0.0f;
      #pragma unroll
      for (int j = 0; j < 16; ++j)
        if (j < i) s = fmaf(-Aw[i][j], y[j], s);
      y[i] = betaS[i] * s;
    }
    float* tm = TmG + (size_t)idx * 256;
    #pragma unroll
    for (int i = 0; i < 16; ++i) tm[i * 16 + ln] = y[i];
  }
}

// ---------------------------------------------------------------------------
// Kernel 3/5: C(N,M) = A(N,K) @ W(M,K)^T, f16 operands, f32 accumulate.
// 256 threads = 8 waves arranged 4x2; block tile 128x128; wave tile 32x64
// (2x4 accumulators -> 8 WMMA per 12 b128 loads per K-step).
// ---------------------------------------------------------------------------
__global__ __launch_bounds__(256) void gemm_f16(
    const _Float16* __restrict__ A, const _Float16* __restrict__ W,
    float* __restrict__ C, int N, int K, int M)
{
  const int tid = threadIdx.x;
  const int wid = tid >> 5, ln = tid & 31;
  const int nH  = ln & 15,  hi = ln >> 4;
  const int wr  = wid >> 1, wc = wid & 1;
  const long rowBase = (long)blockIdx.x * 128 + wr * 32;
  const int  colBase = blockIdx.y * 128 + wc * 64;

  const _Float16* ap[2];
  const _Float16* wp[4];
  #pragma unroll
  for (int rt = 0; rt < 2; ++rt)
    ap[rt] = A + (rowBase + rt * 16 + nH) * (long)K + hi * 8;
  #pragma unroll
  for (int ct = 0; ct < 4; ++ct)
    wp[ct] = W + (colBase + ct * 16 + nH) * (long)K + hi * 8;

  v8f acc[2][4] = {};

  for (int kb = 0; kb < K; kb += 32) {
    v16h afr[2], bfr[4];
    #pragma unroll
    for (int rt = 0; rt < 2; ++rt) {
      v8h l0 = *(const v8h*)(ap[rt] + kb), l1 = *(const v8h*)(ap[rt] + kb + 16);
      afr[rt] = __builtin_shufflevector(l0, l1,
                  0,1,2,3,4,5,6,7,8,9,10,11,12,13,14,15);
    }
    #pragma unroll
    for (int ct = 0; ct < 4; ++ct) {
      v8h l0 = *(const v8h*)(wp[ct] + kb), l1 = *(const v8h*)(wp[ct] + kb + 16);
      bfr[ct] = __builtin_shufflevector(l0, l1,
                  0,1,2,3,4,5,6,7,8,9,10,11,12,13,14,15);
    }
    __builtin_prefetch(ap[0] + kb + 32, 0, 1);
    __builtin_prefetch(wp[0] + kb + 32, 0, 1);
    #pragma unroll
    for (int rt = 0; rt < 2; ++rt)
      #pragma unroll
      for (int ct = 0; ct < 4; ++ct)
        acc[rt][ct] = wmma_f16(afr[rt], bfr[ct], acc[rt][ct]);
  }

  #pragma unroll
  for (int rt = 0; rt < 2; ++rt)
    #pragma unroll
    for (int ct = 0; ct < 4; ++ct) {
      const int col = colBase + ct * 16 + nH;
      float* cp = C + (rowBase + rt * 16 + hi * 8) * (long)M + col;
      #pragma unroll
      for (int g = 0; g < 8; ++g) cp[(long)g * M] = acc[rt][ct][g];
    }
}

// ---------------------------------------------------------------------------
// Kernel 4: serial chunked delta-rule recurrence with double-buffered async
// staging. One block per (b,h); 4 waves; dynamic LDS, manual layout.
// Per chunk: U = V - W*S ; err = Tm*U ; O = (R*S + Mm*err)*gate ; S += W^T*err
// Chunk ci+1's tiles stream in via ASYNCcnt while chunk ci computes.
// ---------------------------------------------------------------------------
#define SL_F   0                              // 64x68 state
#define WC0_F  (SL_F + DHEAD * LSTRIDE)       // 4352
#define RC0_F  (WC0_F + CHUNK * LSTRIDE)      // 5440
#define UC0_F  (RC0_F + CHUNK * LSTRIDE)      // 6528
#define WC1_F  (UC0_F + CHUNK * LSTRIDE)      // 7616
#define RC1_F  (WC1_F + CHUNK * LSTRIDE)      // 8704
#define UC1_F  (RC1_F + CHUNK * LSTRIDE)      // 9792
#define EC_F   (UC1_F + CHUNK * LSTRIDE)      // 10880
#define TM0_F  (EC_F  + CHUNK * LSTRIDE)      // 11968
#define MM0_F  (TM0_F + 16 * TSTRIDE)         // 12288
#define TM1_F  (MM0_F + 16 * TSTRIDE)         // 12608
#define MM1_F  (TM1_F + 16 * TSTRIDE)         // 12928
#define GT0_F  (MM1_F + 16 * TSTRIDE)         // 13248
#define GT1_F  (GT0_F + 16)                   // 13264
#define SMEM_F (GT1_F + 16)                   // 13280 floats
#define SMEM_BYTES (SMEM_F * 4)               // 53120 B
#define DBUF_MAIN (WC1_F - WC0_F)             // 3264
#define DBUF_TM   (TM1_F - TM0_F)             // 640
#define DBUF_GT   (GT1_F - GT0_F)             // 16

__device__ __forceinline__ void stage_async(
    int tid, long rowBase, int h, int t0, size_t cidx,
    const float* rk, const float* v, const float* gate,
    const float* TmG, const float* MmG,
    unsigned dMain, unsigned dTm, unsigned dGt)
{
  #pragma unroll
  for (int s = tid; s < 256; s += 128) {        // 16 rows x 16 b128 segs
    const int r = s >> 4, c4 = (s & 15) * 4;
    const long g = (rowBase + t0 + r) * (long)DMODEL + h * DHEAD + c4;
    async_copy_b128((RC0_F + dMain + r * LSTRIDE + c4) * 4, rk + g);
    async_copy_b128((UC0_F + dMain + r * LSTRIDE + c4) * 4, v + g);
    const int tw = t0 + r - 1;                  // wk[t] = rk[t-1]
    if (tw >= 0)
      async_copy_b128((WC0_F + dMain + r * LSTRIDE + c4) * 4,
                      rk + (rowBase + tw) * (long)DMODEL + h * DHEAD + c4);
  }
  if (tid < 64) {                               // Tm & Mm tiles
    const int i = tid >> 2, j4 = (tid & 3) * 4;
    async_copy_b128((TM0_F + dTm + i * TSTRIDE + j4) * 4, TmG + cidx + i * 16 + j4);
    async_copy_b128((MM0_F + dTm + i * TSTRIDE + j4) * 4, MmG + cidx + i * 16 + j4);
  }
  if (tid < 16)                                 // gate scalars (strided)
    async_copy_b32((GT0_F + dGt + tid) * 4,
                   gate + (rowBase + t0 + tid) * HEADS + h);
}

__global__ __launch_bounds__(128) void delta_recurrence(
    const float* __restrict__ rk, const float* __restrict__ v,
    const float* __restrict__ gate,
    const float* __restrict__ TmG, const float* __restrict__ MmG,
    _Float16* __restrict__ oh)
{
  extern __shared__ float smem[];
  float (*Sl)[LSTRIDE] = (float(*)[LSTRIDE])(smem + SL_F);
  float (*Ec)[LSTRIDE] = (float(*)[LSTRIDE])(smem + EC_F);

  const int bh  = blockIdx.x;
  const int b   = bh / HEADS, h = bh % HEADS;
  const int tid = threadIdx.x;
  const int wid = tid >> 5, ln = tid & 31;
  const int nH  = ln & 15,  hi = ln >> 4;
  const long rowBase = (long)b * T_LEN;

  for (int i = tid; i < DHEAD * LSTRIDE; i += 128) smem[SL_F + i] = 0.f;

  // Prologue: stage chunk 0 into buffer 0
  stage_async(tid, rowBase, h, 0, (size_t)(bh * NCHUNK) * 256,
              rk, v, gate, TmG, MmG, 0u, 0u, 0u);

  for (int ci = 0; ci < NCHUNK; ++ci) {
    const int t0 = ci * CHUNK;
    const int pb = ci & 1;
    const unsigned dMain = pb ? (unsigned)DBUF_MAIN : 0u;
    const unsigned dTm   = pb ? (unsigned)DBUF_TM   : 0u;
    const unsigned dGt   = pb ? (unsigned)DBUF_GT   : 0u;
    float (*Wc)[LSTRIDE] = (float(*)[LSTRIDE])(smem + WC0_F + dMain);
    float (*Rc)[LSTRIDE] = (float(*)[LSTRIDE])(smem + RC0_F + dMain);
    float (*Uc)[LSTRIDE] = (float(*)[LSTRIDE])(smem + UC0_F + dMain);
    float (*Tm)[TSTRIDE] = (float(*)[TSTRIDE])(smem + TM0_F + dTm);
    float (*Mm)[TSTRIDE] = (float(*)[TSTRIDE])(smem + MM0_F + dTm);
    const float* Gt = smem + GT0_F + dGt;

    wait_async();                               // this chunk's tiles resident
    if (ci == 0 && tid < 64) Wc[0][tid] = 0.f;  // wk[0] = 0
    __syncthreads();

    // Overlap: stream chunk ci+1 into the other buffers during compute
    if (ci + 1 < NCHUNK)
      stage_async(tid, rowBase, h, t0 + CHUNK,
                  (size_t)(bh * NCHUNK + ci + 1) * 256,
                  rk, v, gate, TmG, MmG,
                  pb ? 0u : (unsigned)DBUF_MAIN,
                  pb ? 0u : (unsigned)DBUF_TM,
                  pb ? 0u : (unsigned)DBUF_GT);

    // ---- U = V - W*S  (wave = 16-wide column strip) ----
    {
      v8f acc = {};
      #pragma unroll
      for (int i = 0; i < 4; ++i)
        #pragma unroll
        for (int kk = 0; kk < 4; ++kk) {
          const int kb = i * 16 + kk * 4;
          v2f a, bf;
          a.x  = Wc[nH][kb + hi * 2 + 0];
          a.y  = Wc[nH][kb + hi * 2 + 1];
          bf.x = Sl[kb + hi * 2 + 0][wid * 16 + nH];
          bf.y = Sl[kb + hi * 2 + 1][wid * 16 + nH];
          acc = wmma_f32(a, bf, acc);
        }
      const int col = wid * 16 + nH, rb = hi * 8;
      #pragma unroll
      for (int g = 0; g < 8; ++g) Uc[rb + g][col] -= acc[g];
    }
    __syncthreads();

    // ---- err = Tm * U ----
    {
      v8f acc = {};
      #pragma unroll
      for (int kk = 0; kk < 4; ++kk) {
        const int kb = kk * 4;
        v2f a, bf;
        a.x  = Tm[nH][kb + hi * 2 + 0];
        a.y  = Tm[nH][kb + hi * 2 + 1];
        bf.x = Uc[kb + hi * 2 + 0][wid * 16 + nH];
        bf.y = Uc[kb + hi * 2 + 1][wid * 16 + nH];
        acc = wmma_f32(a, bf, acc);
      }
      const int col = wid * 16 + nH, rb = hi * 8;
      #pragma unroll
      for (int g = 0; g < 8; ++g) Ec[rb + g][col] = acc[g];
    }
    __syncthreads();

    // ---- O = (R*S + Mm*err) * gate  -> f16 global ----
    {
      v8f acc = {};
      #pragma unroll
      for (int i = 0; i < 4; ++i)
        #pragma unroll
        for (int kk = 0; kk < 4; ++kk) {
          const int kb = i * 16 + kk * 4;
          v2f a, bf;
          a.x  = Rc[nH][kb + hi * 2 + 0];
          a.y  = Rc[nH][kb + hi * 2 + 1];
          bf.x = Sl[kb + hi * 2 + 0][wid * 16 + nH];
          bf.y = Sl[kb + hi * 2 + 1][wid * 16 + nH];
          acc = wmma_f32(a, bf, acc);
        }
      #pragma unroll
      for (int kk = 0; kk < 4; ++kk) {
        const int k0 = kk * 4 + hi * 2, k1 = k0 + 1;
        v2f a, bf;
        a.x  = Mm[nH][k0];                      // premasked lower-incl
        a.y  = Mm[nH][k1];
        bf.x = Ec[k0][wid * 16 + nH];
        bf.y = Ec[k1][wid * 16 + nH];
        acc = wmma_f32(a, bf, acc);
      }
      const int col = wid * 16 + nH, rb = hi * 8;
      #pragma unroll
      for (int g = 0; g < 8; ++g) {
        const int r = rb + g;
        oh[(rowBase + t0 + r) * (long)DMODEL + h * DHEAD + col] =
            (_Float16)(acc[g] * Gt[r]);
      }
    }
    __syncthreads();

    // ---- S += W^T * err ----  (no trailing barrier: next iteration's
    //      post-wait barrier orders these writes before the next U stage)
    #pragma unroll
    for (int i = 0; i < 4; ++i) {
      v8f acc = {};
      #pragma unroll
      for (int kk = 0; kk < 4; ++kk) {
        const int k0 = kk * 4 + hi * 2, k1 = k0 + 1;
        v2f a, bf;
        a.x  = Wc[k0][i * 16 + nH];             // (W^T)[m][k] = Wc[k][m]
        a.y  = Wc[k1][i * 16 + nH];
        bf.x = Ec[k0][wid * 16 + nH];
        bf.y = Ec[k1][wid * 16 + nH];
        acc = wmma_f32(a, bf, acc);
      }
      const int col = wid * 16 + nH, rb = hi * 8;
      #pragma unroll
      for (int g = 0; g < 8; ++g) Sl[i * 16 + rb + g][col] += acc[g];
    }
  }
}

// ---------------------------------------------------------------------------
// Workspace: rk 64MB | v 64MB | gate 1MB | beta 1MB | Tm 16MB | Mm 16MB |
//            xh 32MB | oh 32MB | Wvh 2MB | Woh 2MB   (~230MB)
// ---------------------------------------------------------------------------
extern "C" void kernel_launch(void* const* d_in, const int* in_sizes, int n_in,
                              void* d_out, int out_size, void* d_ws, size_t ws_size,
                              hipStream_t stream) {
  const float* x  = (const float*)d_in[0];
  const float* Wv = (const float*)d_in[1];
  const float* Wo = (const float*)d_in[2];
  const float* Wg = (const float*)d_in[3];
  const float* Wb = (const float*)d_in[4];
  float* out = (float*)d_out;

  float* rk  = (float*)d_ws;
  float* vws = rk  + (size_t)NROWS * DMODEL;
  float* gws = vws + (size_t)NROWS * DMODEL;
  float* bws = gws + (size_t)NROWS * HEADS;
  float* TmG = bws + (size_t)NROWS * HEADS;
  float* MmG = TmG + (size_t)64 * NCHUNK * 256;
  _Float16* xh  = (_Float16*)(MmG + (size_t)64 * NCHUNK * 256);
  _Float16* oh  = xh  + (size_t)NROWS * DMODEL;
  _Float16* Wvh = oh  + (size_t)NROWS * DMODEL;
  _Float16* Woh = Wvh + (size_t)DMODEL * DMODEL;

  const long nx4 = (long)NROWS * DMODEL / 4;
  const long nw4 = (long)DMODEL * DMODEL / 4;
  cast_f16_kernel<<<(nx4 + 255) / 256, 256, 0, stream>>>(x,  xh,  nx4);
  cast_f16_kernel<<<(nw4 + 255) / 256, 256, 0, stream>>>(Wv, Wvh, nw4);
  cast_f16_kernel<<<(nw4 + 255) / 256, 256, 0, stream>>>(Wo, Woh, nw4);

  prep_kernel<<<NROWS, 512, 0, stream>>>(x, Wg, Wb, rk, gws, bws);

  chunk_precompute<<<64 * NCHUNK, 32, 0, stream>>>(rk, bws, TmG, MmG);

  dim3 gg(NROWS / 128, DMODEL / 128);
  gemm_f16<<<gg, 256, 0, stream>>>(xh, Wvh, vws, NROWS, DMODEL, DMODEL);

  delta_recurrence<<<BB * HEADS, 128, SMEM_BYTES, stream>>>(
      rk, vws, gws, TmG, MmG, oh);

  gemm_f16<<<gg, 256, 0, stream>>>(oh, Woh, out, NROWS, DMODEL, DMODEL);
}
